// GnnClassifier_26439818674553
// MI455X (gfx1250) — matrix operand order repaired and checked by reading, not compiled, not run on hardware
//
#include <hip/hip_runtime.h>
#include <hip/hip_bf16.h>

// MI455X (gfx1250) GNN classifier — round 2.
// f16 WMMA 16x16x32 (f32 accum) for all GEMMs; h kept f32 (residual) + f16
// mirror (L2-resident gather operand). Weights pre-transposed [N][K] f16 so
// B-fragments are contiguous 32B reads. Edge MLP now processes TWO 16-row
// M-tiles per wave (32 edges) so each B fragment feeds 2 WMMAs, and B loads
// are double-buffered to overlap with WMMA issue. Hardware v_tanh_f32 via
// __builtin_amdgcn_tanhf when available. Workspace need: ~94 MB.

#define NN 50000
#define NE 800000
#define HD 128

typedef __attribute__((ext_vector_type(16))) _Float16 v16h;
typedef __attribute__((ext_vector_type(8)))  float    v8f;
typedef __attribute__((ext_vector_type(4)))  unsigned int u32x4;
typedef __attribute__((ext_vector_type(4)))  float    f32x4;

union V16H { v16h v; u32x4 q[2]; _Float16 h[16]; };

#if __has_builtin(__builtin_amdgcn_tanhf)
#define TANHF(x) __builtin_amdgcn_tanhf(x)
#else
#define TANHF(x) tanhf(x)
#endif

// 16 contiguous f16 (two 16B loads) -> B fragment (lane = col N, 16 consecutive K)
static __device__ __forceinline__ v16h ldB(const _Float16* p) {
  V16H u;
  u.q[0] = *(const u32x4*)(p);
  u.q[1] = *(const u32x4*)(p + 8);
  return u.v;
}

// A fragment: row base pointer (already at +k0), chunks at {2*off, 2*off+8}, off = hiH*8
static __device__ __forceinline__ v16h ldA(const _Float16* p, int off) {
  V16H u;
  u.q[0] = *(const u32x4*)(p + 2 * off);
  u.q[1] = *(const u32x4*)(p + 2 * off + 8);
  return u.v;
}

static __device__ __forceinline__ v8f wmma16(v16h a, v16h b, v8f c) {
  return __builtin_amdgcn_wmma_f32_16x16x32_f16(false, a, false, b, (short)0, c,
                                                false, false);
}

static __device__ __forceinline__ float sigmoidf_(float x) {
  return 1.0f / (1.0f + __expf(-x));
}

static __device__ __forceinline__ void bias_init(v8f* c, const float* bias, int col) {
#pragma unroll
  for (int j = 0; j < 8; ++j) {
    float bj = bias[j * 16 + col];
#pragma unroll
    for (int v = 0; v < 8; ++v) c[j][v] = bj;
  }
}

static __device__ __forceinline__ void store_tanh(_Float16* tile, const v8f* c,
                                                  int col, int offA) {
#pragma unroll
  for (int j = 0; j < 8; ++j)
#pragma unroll
    for (int v = 0; v < 8; ++v)
      tile[(v + offA) * HD + j * 16 + col] = (_Float16)TANHF(c[j][v]);
}

// ---- chained 128->128 tanh layer on TWO per-wave LDS tiles [16][128] f16.
// Each B fragment (double-buffered) feeds both tiles' WMMAs.
static __device__ __forceinline__ void layer128_2(_Float16* t0, _Float16* t1,
                                                  const _Float16* __restrict__ WT,
                                                  const float* __restrict__ bias,
                                                  int col, int hiH, int offA) {
  v8f c0[8], c1[8];
  bias_init(c0, bias, col);
  bias_init(c1, bias, col);
#pragma unroll
  for (int ks = 0; ks < 4; ++ks) {
    const int k0 = ks * 32;
    v16h a0 = ldA(t0 + col * HD + k0, offA);
    v16h a1 = ldA(t1 + col * HD + k0, offA);
    v16h b = ldB(WT + col * HD + k0 + hiH * 16);
#pragma unroll
    for (int j = 0; j < 8; ++j) {
      v16h bn = b;
      if (j < 7) bn = ldB(WT + ((j + 1) * 16 + col) * HD + k0 + hiH * 16);
      c0[j] = wmma16(a0, b, c0[j]);
      c1[j] = wmma16(a1, b, c1[j]);
      b = bn;
    }
  }
  // per-wave in-order LDS: reads above issue before writes below (WAR safe)
  store_tanh(t0, c0, col, offA);
  store_tanh(t1, c1, col, offA);
  asm volatile("s_wait_dscnt 0" ::: "memory");  // cross-lane LDS RAW
}

// ======================= edge MLP ==========================================
// Per wave: 32 edges (two 16-row tiles). A = [h16[src] | h16[dst]] (K=256),
// then 2x 128x128 tanh layers, then 128->1 dot + sigmoid.
__global__ __launch_bounds__(256) void edge_mlp_kernel(
    const long long* __restrict__ esrc, const long long* __restrict__ edst,
    const _Float16* __restrict__ h16,
    const _Float16* __restrict__ We1T, const float* __restrict__ be1,
    const _Float16* __restrict__ We2T, const float* __restrict__ be2,
    const _Float16* __restrict__ We3T, const float* __restrict__ be3,
    const _Float16* __restrict__ We4T, const float* __restrict__ be4,
    float* __restrict__ eout) {
  __shared__ _Float16 act[8][2][16 * HD];  // 64 KB, two tiles per wave
  const int lane = threadIdx.x & 31;
  const int wave = threadIdx.x >> 5;
  const int col  = lane & 15;
  const int hiH  = lane >> 4;
  const int offA = hiH * 8;
  const long long e0 = ((long long)blockIdx.x * 8 + wave) * 32;  // E%256==0

  const int s0 = (int)esrc[e0 + col];
  const int d0 = (int)edst[e0 + col];
  const int s1 = (int)esrc[e0 + 16 + col];
  const int d1 = (int)edst[e0 + 16 + col];
  const _Float16* hs0 = h16 + (long long)s0 * HD;
  const _Float16* hd0 = h16 + (long long)d0 * HD;
  const _Float16* hs1 = h16 + (long long)s1 * HD;
  const _Float16* hd1 = h16 + (long long)d1 * HD;

  // layer 1: [16x256] x [256x128], two M-tiles
  v8f c0[8], c1[8];
  bias_init(c0, be1, col);
  bias_init(c1, be1, col);
#pragma unroll
  for (int ks = 0; ks < 8; ++ks) {
    const int k0 = ks * 32;
    const _Float16* p0 = (k0 < HD ? hs0 : hd0) + (k0 & (HD - 1));
    const _Float16* p1 = (k0 < HD ? hs1 : hd1) + (k0 & (HD - 1));
    v16h a0 = ldA(p0, offA);
    v16h a1 = ldA(p1, offA);
    v16h b = ldB(We1T + col * 256 + k0 + hiH * 16);
#pragma unroll
    for (int j = 0; j < 8; ++j) {
      v16h bn = b;
      if (j < 7) bn = ldB(We1T + ((j + 1) * 16 + col) * 256 + k0 + hiH * 16);
      c0[j] = wmma16(a0, b, c0[j]);
      c1[j] = wmma16(a1, b, c1[j]);
      b = bn;
    }
  }
  _Float16* t0 = act[wave][0];
  _Float16* t1 = act[wave][1];
  store_tanh(t0, c0, col, offA);
  store_tanh(t1, c1, col, offA);
  asm volatile("s_wait_dscnt 0" ::: "memory");

  layer128_2(t0, t1, We2T, be2, col, hiH, offA);
  layer128_2(t0, t1, We3T, be3, col, hiH, offA);

  // layer 4: 128 -> 1 dot; lane pair (L, L+16) split K halves
  float s0f = 0.f, s1f = 0.f;
  const int kb = hiH * 64;
#pragma unroll
  for (int kk = 0; kk < 64; kk += 8) {
    V16H wa, ta0, ta1;
    wa.q[0]  = *(const u32x4*)&We4T[kb + kk];
    ta0.q[0] = *(const u32x4*)&t0[col * HD + kb + kk];
    ta1.q[0] = *(const u32x4*)&t1[col * HD + kb + kk];
#pragma unroll
    for (int i = 0; i < 8; ++i) {
      s0f += (float)ta0.h[i] * (float)wa.h[i];
      s1f += (float)ta1.h[i] * (float)wa.h[i];
    }
  }
  s0f += __shfl_xor(s0f, 16, 32);
  s1f += __shfl_xor(s1f, 16, 32);
  if (!hiH) {
    const float b4 = be4[0];
    eout[e0 + col]      = sigmoidf_(s0f + b4);
    eout[e0 + 16 + col] = sigmoidf_(s1f + b4);
  }
}

// ======================= node MLP ==========================================
static __device__ __forceinline__ v16h ldA_f32(const float* __restrict__ mi,
                                               const float* __restrict__ mo,
                                               const float* __restrict__ hp,
                                               int node, int k0, int off) {
  V16H u;
#pragma unroll
  for (int ch = 0; ch < 2; ++ch) {
    const int kg = k0 + ch * 8 + 2 * off;
    const float* src;
    int kl;
    if (kg < 128)      { src = mi; kl = kg; }
    else if (kg < 256) { src = mo; kl = kg - 128; }
    else               { src = hp; kl = kg - 256; }
    const float* p = src + (long long)node * HD + kl;
    f32x4 a = *(const f32x4*)p;
    f32x4 b = *(const f32x4*)(p + 4);
#pragma unroll
    for (int i = 0; i < 4; ++i) {
      u.h[ch * 8 + i]     = (_Float16)a[i];
      u.h[ch * 8 + 4 + i] = (_Float16)b[i];
    }
  }
  return u.v;
}

// single-tile chained layer (node path)
static __device__ __forceinline__ void layer128_1(_Float16* tile,
                                                  const _Float16* __restrict__ WT,
                                                  const float* __restrict__ bias,
                                                  int col, int hiH, int offA) {
  v8f c[8];
  bias_init(c, bias, col);
#pragma unroll
  for (int ks = 0; ks < 4; ++ks) {
    const int k0 = ks * 32;
    v16h a = ldA(tile + col * HD + k0, offA);
    v16h b = ldB(WT + col * HD + k0 + hiH * 16);
#pragma unroll
    for (int j = 0; j < 8; ++j) {
      v16h bn = b;
      if (j < 7) bn = ldB(WT + ((j + 1) * 16 + col) * HD + k0 + hiH * 16);
      c[j] = wmma16(a, b, c[j]);
      b = bn;
    }
  }
  store_tanh(tile, c, col, offA);
  asm volatile("s_wait_dscnt 0" ::: "memory");
}

__global__ __launch_bounds__(256) void node_mlp_kernel(
    const float* __restrict__ mi, const float* __restrict__ mo,
    float* __restrict__ h, _Float16* __restrict__ h16,
    const _Float16* __restrict__ Wn1T, const float* __restrict__ bn1,
    const _Float16* __restrict__ Wn2T, const float* __restrict__ bn2,
    const _Float16* __restrict__ Wn3T, const float* __restrict__ bn3,
    const _Float16* __restrict__ Wn4T, const float* __restrict__ bn4) {
  __shared__ _Float16 act[8][16 * HD];
  const int lane = threadIdx.x & 31;
  const int wave = threadIdx.x >> 5;
  const int col  = lane & 15;
  const int hiH  = lane >> 4;
  const int offA = hiH * 8;
  const int tid  = blockIdx.x * 8 + wave;      // node tile
  if (tid * 16 >= NN) return;                  // wave-uniform exit (no barriers used)
  const int n0 = tid * 16;
  const int node = n0 + col;

  // layer 1: [16x384] x [384x128]
  v8f c[8];
  bias_init(c, bn1, col);
#pragma unroll
  for (int ks = 0; ks < 12; ++ks) {
    const int k0 = ks * 32;
    v16h a = ldA_f32(mi, mo, h, node, k0, offA);
    v16h b = ldB(Wn1T + col * 384 + k0 + hiH * 16);
#pragma unroll
    for (int j = 0; j < 8; ++j) {
      v16h bn = b;
      if (j < 7) bn = ldB(Wn1T + ((j + 1) * 16 + col) * 384 + k0 + hiH * 16);
      c[j] = wmma16(a, b, c[j]);
      b = bn;
    }
  }
  _Float16* tile = act[wave];
  store_tanh(tile, c, col, offA);
  asm volatile("s_wait_dscnt 0" ::: "memory");

  layer128_1(tile, Wn2T, bn2, col, hiH, offA);
  layer128_1(tile, Wn3T, bn3, col, hiH, offA);

  // layer 4 + residual
  v8f c4[8];
  bias_init(c4, bn4, col);
#pragma unroll
  for (int ks = 0; ks < 4; ++ks) {
    const int k0 = ks * 32;
    v16h a = ldA(tile + col * HD + k0, offA);
    v16h b = ldB(Wn4T + col * HD + k0 + hiH * 16);
#pragma unroll
    for (int j = 0; j < 8; ++j) {
      v16h bn = b;
      if (j < 7) bn = ldB(Wn4T + ((j + 1) * 16 + col) * HD + k0 + hiH * 16);
      c4[j] = wmma16(a, b, c4[j]);
      b = bn;
    }
  }
#pragma unroll
  for (int j = 0; j < 8; ++j)
#pragma unroll
    for (int v = 0; v < 8; ++v) {
      const int m = v + offA;
      const long long idx = (long long)(n0 + m) * HD + j * 16 + col;
      float nh = h[idx] + TANHF(c4[j][v]);
      h[idx]   = nh;
      h16[idx] = (_Float16)nh;
    }
}

// ======================= small kernels =====================================
__global__ void init_h_kernel(const float* __restrict__ x,
                              const float* __restrict__ Wi,
                              const float* __restrict__ bi,
                              float* __restrict__ h, _Float16* __restrict__ h16) {
  long long i = (long long)blockIdx.x * 256 + threadIdx.x;
  if (i >= (long long)NN * HD) return;
  int n = (int)(i / HD), f = (int)(i % HD);
  float s = bi[f] + x[n * 3 + 0] * Wi[0 * HD + f] + x[n * 3 + 1] * Wi[1 * HD + f] +
            x[n * 3 + 2] * Wi[2 * HD + f];
  float v = TANHF(s);
  h[i] = v;
  h16[i] = (_Float16)v;
}

// dst[n*K+k] = (f16) src[k*Nn + n]   (weight transpose + f32->f16)
__global__ void convtr_kernel(const float* __restrict__ src,
                              _Float16* __restrict__ dst, int K, int Nn) {
  int i = blockIdx.x * 256 + threadIdx.x;
  if (i >= K * Nn) return;
  int n = i / K, k = i % K;
  dst[i] = (_Float16)src[k * Nn + n];
}

__global__ void zero2_kernel(float* __restrict__ a, float* __restrict__ b,
                             long long cnt) {
  long long i = (long long)blockIdx.x * 256 + threadIdx.x;
  if (i < cnt) { a[i] = 0.f; b[i] = 0.f; }
}

// mi[dst] += e*h[src]; mo[src] += e*h[dst]   (4 features per thread)
__global__ void scatter_kernel(const long long* __restrict__ esrc,
                               const long long* __restrict__ edst,
                               const float* __restrict__ ev,
                               const float* __restrict__ h,
                               float* __restrict__ mi, float* __restrict__ mo) {
  long long i = (long long)blockIdx.x * 256 + threadIdx.x;
  if (i >= (long long)NE * 32) return;
  int eid = (int)(i >> 5);
  int f = (int)(i & 31) * 4;
  float e = ev[eid];
  int s = (int)esrc[eid], d = (int)edst[eid];
  f32x4 hs = *(const f32x4*)(h + (long long)s * HD + f);
  f32x4 hd = *(const f32x4*)(h + (long long)d * HD + f);
#pragma unroll
  for (int t = 0; t < 4; ++t) {
    atomicAdd(&mi[(long long)d * HD + f + t], e * hs[t]);
    atomicAdd(&mo[(long long)s * HD + f + t], e * hd[t]);
  }
}

// ======================= launch ============================================
extern "C" void kernel_launch(void* const* d_in, const int* in_sizes, int n_in,
                              void* d_out, int out_size, void* d_ws, size_t ws_size,
                              hipStream_t stream) {
  (void)in_sizes; (void)n_in; (void)out_size; (void)ws_size;
  const float*     x    = (const float*)d_in[0];
  const long long* eidx = (const long long*)d_in[1];
  const long long* esrc = eidx;
  const long long* edst = eidx + NE;
  const float* Wi  = (const float*)d_in[2];  const float* bi  = (const float*)d_in[3];
  const float* We1 = (const float*)d_in[4];  const float* be1 = (const float*)d_in[5];
  const float* We2 = (const float*)d_in[6];  const float* be2 = (const float*)d_in[7];
  const float* We3 = (const float*)d_in[8];  const float* be3 = (const float*)d_in[9];
  const float* We4 = (const float*)d_in[10]; const float* be4 = (const float*)d_in[11];
  const float* Wn1 = (const float*)d_in[12]; const float* bn1 = (const float*)d_in[13];
  const float* Wn2 = (const float*)d_in[14]; const float* bn2 = (const float*)d_in[15];
  const float* Wn3 = (const float*)d_in[16]; const float* bn3 = (const float*)d_in[17];
  const float* Wn4 = (const float*)d_in[18]; const float* bn4 = (const float*)d_in[19];

  size_t off = 0;
  auto take = [&](size_t bytes) -> void* {
    void* p = (char*)d_ws + off;
    off += (bytes + 255) & ~(size_t)255;
    return p;
  };
  float*     h    = (float*)take(sizeof(float) * NN * HD);
  _Float16*  h16  = (_Float16*)take(sizeof(_Float16) * NN * HD);
  float*     mi   = (float*)take(sizeof(float) * NN * HD);
  float*     mo   = (float*)take(sizeof(float) * NN * HD);
  float*     ebuf = (float*)take(sizeof(float) * NE);
  _Float16*  We1T = (_Float16*)take(sizeof(_Float16) * 256 * 128);
  _Float16*  We2T = (_Float16*)take(sizeof(_Float16) * 128 * 128);
  _Float16*  We3T = (_Float16*)take(sizeof(_Float16) * 128 * 128);
  _Float16*  We4T = (_Float16*)take(sizeof(_Float16) * 128);
  _Float16*  Wn1T = (_Float16*)take(sizeof(_Float16) * 384 * 128);
  _Float16*  Wn2T = (_Float16*)take(sizeof(_Float16) * 128 * 128);
  _Float16*  Wn3T = (_Float16*)take(sizeof(_Float16) * 128 * 128);
  _Float16*  Wn4T = (_Float16*)take(sizeof(_Float16) * 128 * 128);

  const dim3 blk(256);
  auto cdiv = [](long long a, long long b) { return (unsigned)((a + b - 1) / b); };

  convtr_kernel<<<cdiv(256 * 128, 256), blk, 0, stream>>>(We1, We1T, 256, 128);
  convtr_kernel<<<cdiv(128 * 128, 256), blk, 0, stream>>>(We2, We2T, 128, 128);
  convtr_kernel<<<cdiv(128 * 128, 256), blk, 0, stream>>>(We3, We3T, 128, 128);
  convtr_kernel<<<1, blk, 0, stream>>>(We4, We4T, 128, 1);
  convtr_kernel<<<cdiv(384 * 128, 256), blk, 0, stream>>>(Wn1, Wn1T, 384, 128);
  convtr_kernel<<<cdiv(128 * 128, 256), blk, 0, stream>>>(Wn2, Wn2T, 128, 128);
  convtr_kernel<<<cdiv(128 * 128, 256), blk, 0, stream>>>(Wn3, Wn3T, 128, 128);
  convtr_kernel<<<cdiv(128 * 128, 256), blk, 0, stream>>>(Wn4, Wn4T, 128, 128);

  init_h_kernel<<<cdiv((long long)NN * HD, 256), blk, 0, stream>>>(x, Wi, bi, h, h16);

  const unsigned edge_blocks = NE / 256;              // 3125, exact (32 edges/wave)
  const unsigned node_blocks = cdiv((NN + 15) / 16, 8);

  for (int layer = 0; layer < 4; ++layer) {
    edge_mlp_kernel<<<edge_blocks, blk, 0, stream>>>(
        esrc, edst, h16, We1T, be1, We2T, be2, We3T, be3, We4T, be4, ebuf);
    zero2_kernel<<<cdiv((long long)NN * HD, 256), blk, 0, stream>>>(mi, mo,
                                                                    (long long)NN * HD);
    scatter_kernel<<<cdiv((long long)NE * 32, 256), blk, 0, stream>>>(esrc, edst,
                                                                      ebuf, h, mi, mo);
    node_mlp_kernel<<<node_blocks, blk, 0, stream>>>(mi, mo, h, h16, Wn1T, bn1,
                                                     Wn2T, bn2, Wn3T, bn3, Wn4T, bn4);
  }
  edge_mlp_kernel<<<edge_blocks, blk, 0, stream>>>(
      esrc, edst, h16, We1T, be1, We2T, be2, We3T, be3, We4T, be4, (float*)d_out);
}
